// KFCore_13099650252820
// MI455X (gfx1250) — matrix operand order repaired and checked by reading, compiled, not run on hardware
//
#include <hip/hip_runtime.h>
#include <hip/hip_bf16.h>
#include <math.h>

typedef _Float16 half_t;
typedef __attribute__((ext_vector_type(16))) _Float16 v16h;
typedef __attribute__((ext_vector_type(8)))  float    v8f;
typedef __attribute__((ext_vector_type(4)))  unsigned int v4u;
typedef __attribute__((ext_vector_type(4)))  unsigned int su4;
typedef __attribute__((ext_vector_type(8)))  unsigned int su8;

#define B_   2
#define S_   2048
#define M_   4096
#define SD_  256
#define MD_  256
#define DH_  64
#define NSLICE 4
#define SLICE_M (M_ / NSLICE)   /* 1024 */
#define MT   32                 /* keys per m-tile */
#define NT_  (SLICE_M / MT)     /* 32 m-tiles per slice */
#define ROWS_WG 64              /* query rows per workgroup (4 waves x 16) */
#define WAVES_WG 4
#define BETA_ 0.125f            /* 1/sqrt(64) */
#define EPS_  1e-7f

#define KRS 72   /* halves stride, row-major K/V tiles in LDS (TDM pad: 128B row + 16B) */
#define VTS 40   /* halves stride, transposed V tile */
#define WLS 40   /* halves stride, per-wave w scratch */

/* ---------------- projection + expmap0 kernels ---------------- */

__global__ __launch_bounds__(64)
void kf_preQ(const float* __restrict__ q, const float* __restrict__ Wq,
             const float* __restrict__ bq,
             half_t* __restrict__ Qr, float* __restrict__ Q0,
             float* __restrict__ qq, float* __restrict__ Qf)
{
    __shared__ float row[SD_];
    __shared__ float red[DH_];
    const int r = blockIdx.x;      /* 0 .. B*S-1 */
    const int t = threadIdx.x;     /* 0 .. 63    */
    for (int i = t; i < SD_; i += 64) row[i] = q[(size_t)r * SD_ + i];
    __syncthreads();
    float acc = bq[t];
    const float* w = Wq + (size_t)t * SD_;
    for (int d = 0; d < SD_; ++d) acc = fmaf(row[d], w[d], acc);
    red[t] = acc * acc; __syncthreads();
    for (int s = 32; s > 0; s >>= 1) { if (t < s) red[t] += red[t + s]; __syncthreads(); }
    const float n = sqrtf(red[0]);
    __syncthreads();
    const float factor = fminf(3.5f / (n + 1e-7f), 1.0f);
    const float x  = acc * factor;
    const float xn = fmaxf(n * factor, EPS_);
    const float c0 = coshf(xn);
    const float xr = sinhf(xn) * x / xn;
    Qr[(size_t)r * DH_ + t] = (half_t)xr;
    Qf[(size_t)r * (DH_ + 1) + 1 + t] = xr;
    red[t] = xr * xr; __syncthreads();
    for (int s = 32; s > 0; s >>= 1) { if (t < s) red[t] += red[t + s]; __syncthreads(); }
    if (t == 0) { Q0[r] = c0; Qf[(size_t)r * (DH_ + 1)] = c0; qq[r] = red[0] - c0 * c0; }
}

__global__ __launch_bounds__(64)
void kf_preK(const float* __restrict__ k, const float* __restrict__ Wk,
             const float* __restrict__ bk,
             half_t* __restrict__ Kr, float* __restrict__ K0)
{
    __shared__ float row[MD_];
    __shared__ float red[DH_];
    const int r = blockIdx.x;
    const int t = threadIdx.x;
    for (int i = t; i < MD_; i += 64) row[i] = k[(size_t)r * MD_ + i];
    __syncthreads();
    float acc = bk[t];
    const float* w = Wk + (size_t)t * MD_;
    for (int d = 0; d < MD_; ++d) acc = fmaf(row[d], w[d], acc);
    red[t] = acc * acc; __syncthreads();
    for (int s = 32; s > 0; s >>= 1) { if (t < s) red[t] += red[t + s]; __syncthreads(); }
    const float n = sqrtf(red[0]);
    const float factor = fminf(3.5f / (n + 1e-7f), 1.0f);
    const float x  = acc * factor;
    const float xn = fmaxf(n * factor, EPS_);
    Kr[(size_t)r * DH_ + t] = (half_t)(sinhf(xn) * x / xn);
    if (t == 0) K0[r] = coshf(xn);
}

__global__ __launch_bounds__(64)
void kf_preV(const float* __restrict__ v, const float* __restrict__ Wk,
             const float* __restrict__ bk, const float* __restrict__ Wv,
             const float* __restrict__ bv,
             half_t* __restrict__ Vr, float* __restrict__ V0, float* __restrict__ vv)
{
    __shared__ float row[MD_];
    __shared__ float hv[DH_];
    __shared__ float red[DH_];
    const int r = blockIdx.x;
    const int t = threadIdx.x;
    for (int i = t; i < MD_; i += 64) row[i] = v[(size_t)r * MD_ + i];
    __syncthreads();
    float a = bk[t];
    const float* w = Wk + (size_t)t * MD_;
    for (int d = 0; d < MD_; ++d) a = fmaf(row[d], w[d], a);
    hv[t] = a; __syncthreads();
    float acc = bv[t];
    const float* w2 = Wv + (size_t)t * DH_;
    for (int d = 0; d < DH_; ++d) acc = fmaf(hv[d], w2[d], acc);
    red[t] = acc * acc; __syncthreads();
    for (int s = 32; s > 0; s >>= 1) { if (t < s) red[t] += red[t + s]; __syncthreads(); }
    const float n = sqrtf(red[0]);
    __syncthreads();
    const float factor = fminf(3.5f / (n + 1e-7f), 1.0f);
    const float x  = acc * factor;
    const float xn = fmaxf(n * factor, EPS_);
    const float c0 = coshf(xn);
    const float xr = sinhf(xn) * x / xn;
    Vr[(size_t)r * DH_ + t] = (half_t)xr;
    red[t] = xr * xr; __syncthreads();
    for (int s = 32; s > 0; s >>= 1) { if (t < s) red[t] += red[t + s]; __syncthreads(); }
    if (t == 0) { V0[r] = c0; vv[r] = red[0] - c0 * c0; }
}

/* ---------------- Tensor Data Mover: 32x128B tile -> LDS with 16B row pad -- */
/* D# group0: count=1 | lds_addr | global_addr[56:0] | type=2                  */
/* D# group1: data_size=8B, pad_enable, pad_interval=32 DWORDs (128B),         */
/*            pad_amount=4 DWORDs (16B)  => LDS row stride 144B == KRS halves  */

__device__ inline void tdm_load_tile(unsigned lds_addr, const void* gptr)
{
    const unsigned long long ga = (unsigned long long)(uintptr_t)gptr;
    su4 g0;
    g0[0] = 1u;                                   /* count=1, user mode        */
    g0[1] = lds_addr;                             /* LDS byte address          */
    g0[2] = (unsigned)ga;                         /* global_addr[31:0]         */
    g0[3] = (unsigned)(ga >> 32) | (2u << 30);    /* global_addr[56:32],type=2 */
    su8 g1;
    g1[0] = (3u << 16)    /* data_size = 8B      */
          | (1u << 20)    /* pad_enable          */
          | (4u << 22)    /* pad_interval: 32 DW */
          | (3u << 25);   /* pad_amount : 4 DW   */
    g1[1] = (16u << 16);  /* tensor_dim0 = 16 (8B units per row)            */
    g1[2] = (32u << 16);  /* tensor_dim1 = 32 rows                          */
    g1[3] = (16u << 16);  /* tile_dim0   = 16                               */
    g1[4] = 32u;          /* tile_dim1   = 32                               */
    g1[5] = 16u;          /* tensor_dim0_stride = 16 (contiguous rows)      */
    g1[6] = 0u;
    g1[7] = 0u;
    asm volatile("tensor_load_to_lds %0, %1" :: "s"(g0), "s"(g1) : "memory");
}

/* ---------------- main flash kernel ------------------------------------- */
/* split over M (NSLICE), online softmax, double-buffered LDS tiles:        */
/* iteration t issues tile t+1's TDM DMA, computes tile t, stages aux for   */
/* t+1, waits TENSORcnt, one barrier per iteration.                         */

union AF { v16h v; v4u q[2]; };
union BF { v16h v; v4u q[2]; };

__global__ __launch_bounds__(128)
void kf_main(const half_t* __restrict__ Qrg, const float* __restrict__ Q0g,
             const float* __restrict__ qqg,
             const half_t* __restrict__ Krg, const float* __restrict__ K0g,
             const half_t* __restrict__ Vrg, const float* __restrict__ V0g,
             const float* __restrict__ vvg,
             float* __restrict__ part)
{
    __shared__ __align__(16) half_t kr_s[2][MT * KRS];
    __shared__ __align__(16) half_t vr_s[2][MT * KRS];
    __shared__ __align__(16) half_t vt_s[2][DH_ * VTS];
    __shared__ __align__(16) half_t w_s[WAVES_WG * 16 * WLS];
    __shared__ float k0_s[2][MT], v0_s[2][MT], vv_s[2][MT];

    const int wg    = blockIdx.x;
    const int slice = wg & (NSLICE - 1);
    const int tmp   = wg >> 2;
    const int sblk  = tmp & (S_ / ROWS_WG - 1);
    const int b     = tmp / (S_ / ROWS_WG);
    const int tid   = threadIdx.x;
    const int wave  = tid >> 5;
    const int lane  = tid & 31;
    const int hi    = lane >> 4;
    const int ln    = lane & 15;
    const int wrow  = sblk * ROWS_WG + wave * 16;
    const size_t qbase  = (size_t)b * S_ + wrow;
    const size_t kvbase = (size_t)b * M_;

    const unsigned lds_kr[2] = { (unsigned)(uintptr_t)&kr_s[0][0],
                                 (unsigned)(uintptr_t)&kr_s[1][0] };
    const unsigned lds_vr[2] = { (unsigned)(uintptr_t)&vr_s[0][0],
                                 (unsigned)(uintptr_t)&vr_s[1][0] };

    /* per-row scalars for this lane's 8 rows (row = e + hi*8) */
    float qq_r[8], q0_r[8];
    for (int e = 0; e < 8; ++e) {
        const int rr = e + hi * 8;
        qq_r[e] = qqg[qbase + rr];
        q0_r[e] = Q0g[qbase + rr];
    }
    /* Q A-fragments (row = wrow + ln), two K-chunks of 32 */
    AF aq[2];
    {
        const v4u* qp = (const v4u*)(Qrg + (qbase + ln) * DH_);
        for (int c = 0; c < 2; ++c) {
            aq[c].q[0] = qp[c * 4 + hi * 2];
            aq[c].q[1] = qp[c * 4 + hi * 2 + 1];
        }
    }

    v8f accv[4];
    float rmax[8], Zs[8], a0[8], ac[8];
    for (int t2 = 0; t2 < 4; ++t2) for (int e = 0; e < 8; ++e) accv[t2][e] = 0.0f;
    for (int e = 0; e < 8; ++e) { rmax[e] = -__builtin_inff(); Zs[e] = 0.f; a0[e] = 0.f; ac[e] = 0.f; }

    /* aux staging: transposed V tile + per-key scalars, all 128 threads */
    auto stage_aux = [&](int m0, int buf) {
        const int r2 = tid >> 2, cg = tid & 3;
        const half_t* src = Vrg + (kvbase + m0 + r2) * DH_ + cg * 16;
        for (int d = 0; d < 16; ++d) vt_s[buf][(cg * 16 + d) * VTS + r2] = src[d];
        if (tid < MT)            k0_s[buf][tid]          = K0g[kvbase + m0 + tid];
        else if (tid < 2 * MT)   v0_s[buf][tid - MT]     = V0g[kvbase + m0 + tid - MT];
        else if (tid < 3 * MT)   vv_s[buf][tid - 2 * MT] = vvg[kvbase + m0 + tid - 2 * MT];
    };

    /* prologue: stage tile 0 into buffer 0 */
    {
        const int m0 = slice * SLICE_M;
        if (wave == 0)      tdm_load_tile(lds_kr[0], Krg + (kvbase + m0) * DH_);
        else if (wave == 1) tdm_load_tile(lds_vr[0], Vrg + (kvbase + m0) * DH_);
        stage_aux(m0, 0);
        if (wave < 2) __builtin_amdgcn_s_wait_tensorcnt(0);
        __syncthreads();
    }

    for (int mt = 0; mt < NT_; ++mt) {
        const int  buf  = mt & 1;
        const int  m0   = slice * SLICE_M + mt * MT;
        const bool more = (mt + 1 < NT_);

        /* kick tile t+1's DMA into the alternate buffer (overlaps compute) */
        if (more) {
            const size_t g1 = (kvbase + m0 + MT) * (size_t)DH_;
            if (wave == 0)      tdm_load_tile(lds_kr[buf ^ 1], Krg + g1);
            else if (wave == 1) tdm_load_tile(lds_vr[buf ^ 1], Vrg + g1);
            if (mt + 2 < NT_)
                __builtin_prefetch((const char*)(Vrg + (kvbase + m0 + 2 * MT) * DH_) + tid * 16, 0, 0);
        }

        float k0c[2], v0c[2], vvc[2];
        for (int t2 = 0; t2 < 2; ++t2) {
            const int j = t2 * 16 + ln;
            k0c[t2] = k0_s[buf][j]; v0c[t2] = v0_s[buf][j]; vvc[t2] = vv_s[buf][j];
        }

        /* sims / ip : rank-1 Minkowski time term in C-init, 2 WMMAs over 64 dims */
        v8f sims[2], ipm[2];
        for (int t2 = 0; t2 < 2; ++t2) {
            const int j = t2 * 16 + ln;
            v8f cs, ci;
            for (int e = 0; e < 8; ++e) { cs[e] = -q0_r[e] * k0c[t2]; ci[e] = -q0_r[e] * v0c[t2]; }
            for (int c = 0; c < 2; ++c) {
                BF bk_, bv_;
                const v4u* kp = (const v4u*)(&kr_s[buf][0] + j * KRS + c * 32 + hi * 16);
                const v4u* vp = (const v4u*)(&vr_s[buf][0] + j * KRS + c * 32 + hi * 16);
                bk_.q[0] = kp[0]; bk_.q[1] = kp[1];
                bv_.q[0] = vp[0]; bv_.q[1] = vp[1];
                cs = __builtin_amdgcn_wmma_f32_16x16x32_f16(false, aq[c].v, false, bk_.v,
                                                            (short)0, cs, false, false);
                ci = __builtin_amdgcn_wmma_f32_16x16x32_f16(false, aq[c].v, false, bv_.v,
                                                            (short)0, ci, false, false);
            }
            sims[t2] = cs; ipm[t2] = ci;
        }

        /* elementwise hyperbolic weights + online softmax */
        float wv[8][2], l0[8][2], nmax[8];
        for (int e = 0; e < 8; ++e) {
            const float la = -BETA_ * sims[0][e];
            const float lb = -BETA_ * sims[1][e];
            l0[e][0] = la; l0[e][1] = lb;
            float tmx = fmaxf(la, lb);
            for (int m2 = 1; m2 < 16; m2 <<= 1) tmx = fmaxf(tmx, __shfl_xor(tmx, m2, 32));
            nmax[e] = fmaxf(rmax[e], tmx);
        }
        for (int e = 0; e < 8; ++e) {
            const float nm = nmax[e];
            const float f  = __expf(rmax[e] - nm);
            rmax[e] = nm;
            float za = 0.f, aa = 0.f, ca = 0.f;
            for (int t2 = 0; t2 < 2; ++t2) {
                const float ip_ = ipm[t2][e];
                const float p   = __expf(l0[e][t2] - nm);
                const float x   = fmaxf(-ip_, 1.0f + EPS_);
                const float dist = acoshf(x);
                const float nom  = vvc[t2] + ip_ * ip_ * (2.0f + qq_r[e]);
                const float g    = dist * rsqrtf(fmaxf(nom, EPS_));
                const float w2   = p * g;
                wv[e][t2] = w2;
                za += p; aa += w2 * v0c[t2]; ca += w2 * ip_;
            }
            Zs[e] = Zs[e] * f + za;
            a0[e] = a0[e] * f + aa;
            ac[e] = ac[e] * f + ca;
            for (int t2 = 0; t2 < 4; ++t2) accv[t2][e] *= f;
        }

        /* bounce w tile through per-wave LDS: C-layout -> A-layout */
        half_t* wl = w_s + wave * 16 * WLS;
        for (int e = 0; e < 8; ++e) {
            const int r2 = e + hi * 8;
            wl[r2 * WLS + ln]      = (half_t)wv[e][0];
            wl[r2 * WLS + 16 + ln] = (half_t)wv[e][1];
        }
        asm volatile("s_wait_dscnt 0" ::: "memory");
        AF aw;
        aw.q[0] = *(const v4u*)(wl + ln * WLS + hi * 8);
        aw.q[1] = *(const v4u*)(wl + ln * WLS + 16 + hi * 8);

        /* accv += w(16x32) @ Vr(32x64) : 4 N-tiles */
        for (int t2 = 0; t2 < 4; ++t2) {
            BF bw;
            const half_t* vp = &vt_s[buf][0] + (t2 * 16 + ln) * VTS + hi * 8;
            bw.q[0] = *(const v4u*)vp;
            bw.q[1] = *(const v4u*)(vp + 16);
            accv[t2] = __builtin_amdgcn_wmma_f32_16x16x32_f16(false, aw.v, false, bw.v,
                                                              (short)0, accv[t2], false, false);
        }

        /* stage tile t+1's aux data, then close the pipeline stage */
        if (more) stage_aux(m0 + MT, buf ^ 1);
        if (more && wave < 2) __builtin_amdgcn_s_wait_tensorcnt(0);
        __syncthreads();
    }

    /* row reductions for the scalar accumulators */
    for (int e = 0; e < 8; ++e) {
        float z2 = Zs[e], a2 = a0[e], c2 = ac[e];
        for (int m2 = 1; m2 < 16; m2 <<= 1) {
            z2 += __shfl_xor(z2, m2, 32);
            a2 += __shfl_xor(a2, m2, 32);
            c2 += __shfl_xor(c2, m2, 32);
        }
        Zs[e] = z2; a0[e] = a2; ac[e] = c2;
    }
    /* write partials: [max, Z, acc0, accc, accr[64]] per (row, slice) */
    for (int e = 0; e < 8; ++e) {
        const int rr = wrow + e + hi * 8;
        const size_t idx = (((size_t)b * S_ + rr) * NSLICE + slice) * 68;
        if (ln == 0) {
            part[idx + 0] = rmax[e]; part[idx + 1] = Zs[e];
            part[idx + 2] = a0[e];   part[idx + 3] = ac[e];
        }
        for (int t2 = 0; t2 < 4; ++t2)
            part[idx + 4 + t2 * 16 + ln] = accv[t2][e];
    }
}

/* ---------------- combine + hyperbolic epilogue ---------------- */

__global__ __launch_bounds__(256)
void kf_combine(const float* __restrict__ part, const float* __restrict__ Qf,
                float* __restrict__ out)
{
    const int row = blockIdx.x * 256 + threadIdx.x;
    if (row >= B_ * S_) return;
    const float* qf = Qf + (size_t)row * (DH_ + 1);
    const size_t base = (size_t)row * NSLICE * 68;
    float gmax = -__builtin_inff();
    for (int s = 0; s < NSLICE; ++s) gmax = fmaxf(gmax, part[base + s * 68]);
    float Zt = 0.f, a0t = 0.f, act = 0.f;
    float accr[64];
    for (int d = 0; d < 64; ++d) accr[d] = 0.f;
    for (int s = 0; s < NSLICE; ++s) {
        const float* p = part + base + s * 68;
        const float f = __expf(p[0] - gmax);
        Zt += f * p[1]; a0t += f * p[2]; act += f * p[3];
        for (int d = 0; d < 64; ++d) accr[d] = fmaf(f, p[4 + d], accr[d]);
    }
    const float inv = 1.0f / Zt;
    const float cw  = act * inv;
    const float q0  = qf[0];
    const float tm0 = a0t * inv + cw * q0;
    float msum = -tm0 * tm0;
    for (int d = 0; d < 64; ++d) {
        const float t = accr[d] * inv + cw * qf[1 + d];
        accr[d] = t;
        msum = fmaf(t, t, msum);
    }
    const float un = sqrtf(fmaxf(msum, EPS_));
    const float ch = coshf(un), si = sinhf(un) / un;
    const float z0 = ch * q0 + si * tm0;
    float yn2 = 0.f;
    for (int d = 0; d < 64; ++d) {
        const float z = ch * qf[1 + d] + si * accr[d];
        accr[d] = z;
        yn2 = fmaf(z, z, yn2);
    }
    const float yn = fmaxf(sqrtf(yn2), EPS_);
    const float dd = acoshf(fmaxf(z0, 1.0f + EPS_));
    const float sc = dd / yn;
    float* o = out + (size_t)row * DH_;
    for (int d = 0; d < 64; ++d) o[d] = sc * accr[d];
}

/* ---------------- launch ---------------- */

extern "C" void kernel_launch(void* const* d_in, const int* in_sizes, int n_in,
                              void* d_out, int out_size, void* d_ws, size_t ws_size,
                              hipStream_t stream) {
    (void)in_sizes; (void)n_in; (void)out_size; (void)ws_size;
    const float* queries = (const float*)d_in[0];
    const float* keys    = (const float*)d_in[1];
    const float* values  = (const float*)d_in[2];
    const float* Wq = (const float*)d_in[3];
    const float* bq = (const float*)d_in[4];
    const float* Wk = (const float*)d_in[5];
    const float* bk = (const float*)d_in[6];
    const float* Wv = (const float*)d_in[7];
    const float* bv = (const float*)d_in[8];
    float* out = (float*)d_out;
    char*  ws  = (char*)d_ws;

    size_t off = 0;
    auto alloc = [&](size_t bytes) { size_t o = off; off = (off + bytes + 255) & ~(size_t)255; return o; };
    const size_t qr_o = alloc((size_t)B_ * S_ * DH_ * 2);
    const size_t q0_o = alloc((size_t)B_ * S_ * 4);
    const size_t qq_o = alloc((size_t)B_ * S_ * 4);
    const size_t qf_o = alloc((size_t)B_ * S_ * (DH_ + 1) * 4);
    const size_t kr_o = alloc((size_t)B_ * M_ * DH_ * 2);
    const size_t k0_o = alloc((size_t)B_ * M_ * 4);
    const size_t vr_o = alloc((size_t)B_ * M_ * DH_ * 2);
    const size_t v0_o = alloc((size_t)B_ * M_ * 4);
    const size_t vv_o = alloc((size_t)B_ * M_ * 4);
    const size_t pt_o = alloc((size_t)B_ * S_ * NSLICE * 68 * 4);

    half_t* Qr = (half_t*)(ws + qr_o);
    float*  Q0 = (float*)(ws + q0_o);
    float*  qq = (float*)(ws + qq_o);
    float*  Qf = (float*)(ws + qf_o);
    half_t* Kr = (half_t*)(ws + kr_o);
    float*  K0 = (float*)(ws + k0_o);
    half_t* Vr = (half_t*)(ws + vr_o);
    float*  V0 = (float*)(ws + v0_o);
    float*  vv = (float*)(ws + vv_o);
    float*  pt = (float*)(ws + pt_o);

    kf_preQ<<<B_ * S_, 64, 0, stream>>>(queries, Wq, bq, Qr, Q0, qq, Qf);
    kf_preK<<<B_ * M_, 64, 0, stream>>>(keys, Wk, bk, Kr, K0);
    kf_preV<<<B_ * M_, 64, 0, stream>>>(values, Wk, bk, Wv, bv, Vr, V0, vv);
    kf_main<<<B_ * (S_ / ROWS_WG) * NSLICE, 128, 0, stream>>>(Qr, Q0, qq, Kr, K0, Vr, V0, vv, pt);
    kf_combine<<<(B_ * S_ + 255) / 256, 256, 0, stream>>>(pt, Qf, out);
}
